// DeLaNFriction_16655883174528
// MI455X (gfx1250) — matrix operand order repaired and checked by reading, not compiled, not run on hardware
//
#include <hip/hip_runtime.h>

typedef _Float16 half_t;
typedef _Float16 v16h __attribute__((ext_vector_type(16)));
typedef _Float16 v8h  __attribute__((ext_vector_type(8)));
typedef float    v8f  __attribute__((ext_vector_type(8)));

#define AS 136            // activation row stride in halves (272B: 16B aligned, bank-conflict free)
#define TRI(i,j) ((i)*((i)+1)/2 + (j))

#define LOG2E 1.4426950408889634f
#define LN2   0.6931471805599453f

// f16 weight workspace layout (offsets in halves), per MLP:
//  Wt_l  = W^T padded  [out_pad][in_pad]   (forward / JVP  B-fragments)
//  Wr_l  = W  padded   [in_pad16][out_pad] (backward       B-fragments)
static constexpr int WT_OFF[4] = {0, 4096, 20480, 36864};      // 128*32, 128*128, 128*128, 32*128
static constexpr int WR_OFF[4] = {40960, 43008, 59392, 75776}; // 16*128, 128*128, 128*128, 128*32
static constexpr int MLP_STRIDE = 79872;                       // halves per MLP

struct Params {
    const float* q;
    const float* qd;
    const float* b[12];      // [mlp*4 + layer]
    const half_t* ws;
    float* out;
};

// ---------------- weight prep: fp32 -> f16, transpose + pad ----------------
__global__ void prep_weights(const float* __restrict__ W, int in, int out,
                             int in_pad, int out_pad, int in_pad16,
                             half_t* __restrict__ wt, half_t* __restrict__ wr) {
    int idx = blockIdx.x * blockDim.x + threadIdx.x;
    int wtN = out_pad * in_pad;
    int wrN = in_pad16 * out_pad;
    if (idx < wtN) {
        int n = idx / in_pad, k = idx % in_pad;
        wt[idx] = (n < out && k < in) ? (half_t)W[k * out + n] : (half_t)0.f;
    } else if (idx < wtN + wrN) {
        int j = idx - wtN;
        int k = j / out_pad, n = j % out_pad;
        wr[j] = (k < in && n < out) ? (half_t)W[k * out + n] : (half_t)0.f;
    }
}

// ---------------- fast branch-free transcendentals (raw HW ops) ----------------
__device__ __forceinline__ float fast_exp(float x) {          // e^x
    return __builtin_amdgcn_exp2f(x * LOG2E);
}
__device__ __forceinline__ float softplus_f(float z) {
    // max(z,0) + ln2 * log2(1 + 2^(-|z|*log2e)) : exp2 arg <= 0, never overflows
    const float t = __builtin_amdgcn_exp2f(-fabsf(z) * LOG2E);
    return fmaxf(z, 0.f) + LN2 * __builtin_amdgcn_logf(1.f + t);
}
__device__ __forceinline__ float sigmoid_f(float z) {
    // rcp saturates correctly: z->+inf => 1, z->-inf => 0
    return __builtin_amdgcn_rcpf(1.f + __builtin_amdgcn_exp2f(-z * LOG2E));
}

// ---------------- WMMA fragment helpers ----------------
__device__ __forceinline__ v16h load16(const half_t* p, int gap) {
    v8h lo = *(const v8h*)(p);
    v8h hh = *(const v8h*)(p + gap);
    v16h r;
#pragma unroll
    for (int e = 0; e < 8; ++e) { r[e] = lo[e]; r[e + 8] = hh[e]; }
    return r;
}

// One 16x16 output tile, K = 32*KC.  A from LDS (stride AS), B from global Wt/Wr.
template<int KC>
__device__ __forceinline__ v8f gemm_tile(const half_t* __restrict__ abuf,
                                         const half_t* __restrict__ wt,
                                         int wstride, int n, int lm, int hi) {
    v8f acc = {0.f, 0.f, 0.f, 0.f, 0.f, 0.f, 0.f, 0.f};
    const half_t* arow = abuf + lm * AS + hi * 16;     // A: lane lm = row M, hi splits K halves
    const half_t* brow = wt + n * wstride + hi * 16;   // B: lane lm = col N, contiguous K
#pragma unroll
    for (int c = 0; c < KC; ++c) {
        v16h a = load16(arow + c * 32, 8);
        v16h b = load16(brow + c * 32, 8);
        acc = __builtin_amdgcn_wmma_f32_16x16x32_f16(false, a, false, b, (short)0, acc,
                                                     false, false);
    }
    return acc;
}

// hidden layer: dst = softplus(src@W + b), optionally save sigmoid(z) in registers
template<int KC, bool SAVE>
__device__ __forceinline__ void layer_hidden(const half_t* src, half_t* dst,
                                             const half_t* wt, int wstride,
                                             const float* __restrict__ bias,
                                             float* sig, int wave, int lane) {
    const int lm = lane & 15, hi = lane >> 4;
    const int n = wave * 16 + lm;
    v8f acc = gemm_tile<KC>(src, wt, wstride, n, lm, hi);
    const float b = bias[n];
#pragma unroll
    for (int r = 0; r < 8; ++r) {
        const int m = r + hi * 8;                  // D: VGPR r -> row r (lanes<16) / r+8
        const float z = acc[r] + b;
        dst[m * AS + n] = (half_t)softplus_f(z);
        if (SAVE) sig[r] = sigmoid_f(z);
    }
}

// JVP / VJP chain layer: dst = sig(z) .* (src @ Wfrag)
template<int KC>
__device__ __forceinline__ void layer_mulsig(const half_t* src, half_t* dst,
                                             const half_t* wt, int wstride,
                                             const float* sig, int wave, int lane) {
    const int lm = lane & 15, hi = lane >> 4;
    const int n = wave * 16 + lm;
    v8f acc = gemm_tile<KC>(src, wt, wstride, n, lm, hi);
#pragma unroll
    for (int r = 0; r < 8; ++r)
        dst[(r + hi * 8) * AS + n] = (half_t)(acc[r] * sig[r]);
}

// write 16x32 f16 buffer from a 16x7 global f32 array (zero padded to K=32)
__device__ __forceinline__ void fill_pad(half_t* dst, const float* __restrict__ src, int tid) {
    for (int idx = tid; idx < 16 * 32; idx += 256) {
        const int row = idx >> 5, col = idx & 31;
        const float v = (col < 7) ? src[row * 7 + col] : 0.f;
        dst[row * AS + col] = (half_t)v;
    }
}

__device__ __forceinline__ const half_t* WTp(const half_t* ws, int mlp, int l) {
    return ws + mlp * MLP_STRIDE + WT_OFF[l];
}
__device__ __forceinline__ const half_t* WRp(const half_t* ws, int mlp, int l) {
    return ws + mlp * MLP_STRIDE + WR_OFF[l];
}

// ---------------- main kernel: 16 samples / workgroup, 8 waves ----------------
__global__ __launch_bounds__(256) void delan_kernel(Params P) {
    __shared__ __align__(16) half_t BUF0[16 * AS];
    __shared__ __align__(16) half_t BUF1[16 * AS];
    __shared__ __align__(16) half_t BUF2[16 * AS];
    __shared__ float cM[16][28];    // mass chol outputs
    __shared__ float tC[16][28];    // JVP tangent of chol outputs
    __shared__ float cF[16][28];    // friction chol outputs
    __shared__ float outP[16];      // potential net output
    __shared__ float gqM[16][8];    // J_m^T g_c
    __shared__ float gqP[16][8];    // grad V
    __shared__ float rhsP[16][8];   // coriolis + friction

    const int tid  = threadIdx.x;
    const int wave = tid >> 5;
    const int lane = tid & 31;
    const int lm   = lane & 15;
    const int hi   = lane >> 4;
    const int s0   = blockIdx.x * 16;
    const float* qg  = P.q  + (size_t)s0 * 7;
    const float* qdg = P.qd + (size_t)s0 * 7;
    const half_t* ws = P.ws;

    float sgm1[8], sgm2[8], sgm3[8];   // mass-MLP sigmoid(z_l) for this wave's n-tile
    float sgp1[8], sgp2[8], sgp3[8];   // potential-MLP sigmoid(z_l)

    // ================= mass forward =================
    fill_pad(BUF0, qg, tid);
    __syncthreads();
    layer_hidden<1, true>(BUF0, BUF1, WTp(ws, 0, 0), 32,  P.b[0], sgm1, wave, lane);
    __syncthreads();
    layer_hidden<4, true>(BUF1, BUF0, WTp(ws, 0, 1), 128, P.b[1], sgm2, wave, lane);
    __syncthreads();
    layer_hidden<4, true>(BUF0, BUF1, WTp(ws, 0, 2), 128, P.b[2], sgm3, wave, lane);
    __syncthreads();
    if (wave < 2) {
        v8f acc = gemm_tile<4>(BUF1, WTp(ws, 0, 3), 128, wave * 16 + lm, lm, hi);
        const int j = wave * 16 + lm;
        if (j < 28) {
            const float b = P.b[3][j];
#pragma unroll
            for (int r = 0; r < 8; ++r) cM[r + hi * 8][j] = acc[r] + b;
        }
    }
    __syncthreads();

    // ================= mass JVP (tangent = qd) =================
    fill_pad(BUF0, qdg, tid);
    __syncthreads();
    layer_mulsig<1>(BUF0, BUF1, WTp(ws, 0, 0), 32,  sgm1, wave, lane);
    __syncthreads();
    layer_mulsig<4>(BUF1, BUF0, WTp(ws, 0, 1), 128, sgm2, wave, lane);
    __syncthreads();
    layer_mulsig<4>(BUF0, BUF1, WTp(ws, 0, 2), 128, sgm3, wave, lane);
    __syncthreads();
    if (wave < 2) {
        v8f acc = gemm_tile<4>(BUF1, WTp(ws, 0, 3), 128, wave * 16 + lm, lm, hi);
        const int j = wave * 16 + lm;
        if (j < 28) {
#pragma unroll
            for (int r = 0; r < 8; ++r) tC[r + hi * 8][j] = acc[r];
        }
    }
    __syncthreads();

    // ================= friction forward =================
    fill_pad(BUF0, qg, tid);
    __syncthreads();
    layer_hidden<1, false>(BUF0, BUF1, WTp(ws, 2, 0), 32,  P.b[8],  nullptr, wave, lane);
    __syncthreads();
    layer_hidden<4, false>(BUF1, BUF0, WTp(ws, 2, 1), 128, P.b[9],  nullptr, wave, lane);
    __syncthreads();
    layer_hidden<4, false>(BUF0, BUF1, WTp(ws, 2, 2), 128, P.b[10], nullptr, wave, lane);
    __syncthreads();
    if (wave < 2) {
        v8f acc = gemm_tile<4>(BUF1, WTp(ws, 2, 3), 128, wave * 16 + lm, lm, hi);
        const int j = wave * 16 + lm;
        if (j < 28) {
            const float b = P.b[11][j];
#pragma unroll
            for (int r = 0; r < 8; ++r) cF[r + hi * 8][j] = acc[r] + b;
        }
    }
    __syncthreads();

    // ================= potential forward =================
    fill_pad(BUF0, qg, tid);
    __syncthreads();
    layer_hidden<1, true>(BUF0, BUF1, WTp(ws, 1, 0), 32,  P.b[4], sgp1, wave, lane);
    __syncthreads();
    layer_hidden<4, true>(BUF1, BUF0, WTp(ws, 1, 1), 128, P.b[5], sgp2, wave, lane);
    __syncthreads();
    layer_hidden<4, true>(BUF0, BUF1, WTp(ws, 1, 2), 128, P.b[6], sgp3, wave, lane);
    __syncthreads();
    if (wave == 0) {
        v8f acc = gemm_tile<4>(BUF1, WTp(ws, 1, 3), 128, lm, lm, hi);
        if (lm == 0) {
            const float b = P.b[7][0];
#pragma unroll
            for (int r = 0; r < 8; ++r) outP[r + hi * 8] = acc[r] + b;
        }
    }
    __syncthreads();

    // ========= scalar stage A: cotangent seeds, coriolis, friction (1 lane/sample) =========
    if (tid < 16) {
        const int m = tid;
        float qd_[7];
        for (int i = 0; i < 7; ++i) qd_[i] = qdg[m * 7 + i];
        float L[28];
        { int k = 0;
          for (int i = 0; i < 7; ++i)
              for (int j = 0; j <= i; ++j, ++k)
                  L[k] = (i == j) ? fast_exp(cM[m][k]) : cM[m][k]; }
        float u[7];
        for (int j = 0; j < 7; ++j) {
            float s = 0.f;
            for (int i = j; i < 7; ++i) s += L[TRI(i, j)] * qd_[i];
            u[j] = s;
        }
        // g_c = d(0.5 qd^T M qd)/d chol  ->  mass VJP seed (BUF0)
        { int k = 0;
          for (int i = 0; i < 7; ++i)
              for (int j = 0; j <= i; ++j, ++k) {
                  float g = qd_[i] * u[j];
                  if (i == j) g *= L[k];
                  BUF0[m * AS + k] = (half_t)g;
              }
          for (int k2 = 28; k2 < 32; ++k2) BUF0[m * AS + k2] = (half_t)0.f; }
        // potential VJP seed (BUF2): sigmoid(out)
        BUF2[m * AS + 0] = (half_t)sigmoid_f(outP[m]);
        for (int k2 = 1; k2 < 32; ++k2) BUF2[m * AS + k2] = (half_t)0.f;
        // coriolis = dL*u + L*(dL^T qd)
        float dL[28];
        { int k = 0;
          for (int i = 0; i < 7; ++i)
              for (int j = 0; j <= i; ++j, ++k) {
                  const float t = tC[m][k];
                  dL[k] = (i == j) ? L[k] * t : t;
              } }
        float w_[7];
        for (int j = 0; j < 7; ++j) {
            float s = 0.f;
            for (int i = j; i < 7; ++i) s += dL[TRI(i, j)] * qd_[i];
            w_[j] = s;
        }
        // friction = Lf (Lf^T qd) + eps_B qd
        float Lf[28];
        { int k = 0;
          for (int i = 0; i < 7; ++i)
              for (int j = 0; j <= i; ++j, ++k)
                  Lf[k] = (i == j) ? fast_exp(cF[m][k]) : cF[m][k]; }
        float v_[7];
        for (int j = 0; j < 7; ++j) {
            float s = 0.f;
            for (int i = j; i < 7; ++i) s += Lf[TRI(i, j)] * qd_[i];
            v_[j] = s;
        }
        for (int i = 0; i < 7; ++i) {
            float s = 1e-3f * qd_[i];
            for (int j = 0; j <= i; ++j)
                s += dL[TRI(i, j)] * u[j] + L[TRI(i, j)] * w_[j] + Lf[TRI(i, j)] * v_[j];
            rhsP[m][i] = s;
        }
    }
    __syncthreads();

    // ================= mass backward (VJP) =================
    layer_mulsig<1>(BUF0, BUF1, WRp(ws, 0, 3), 32,  sgm3, wave, lane);
    __syncthreads();
    layer_mulsig<4>(BUF1, BUF0, WRp(ws, 0, 2), 128, sgm2, wave, lane);
    __syncthreads();
    layer_mulsig<4>(BUF0, BUF1, WRp(ws, 0, 1), 128, sgm1, wave, lane);
    __syncthreads();
    if (wave == 0) {
        v8f acc = gemm_tile<4>(BUF1, WRp(ws, 0, 0), 128, lm, lm, hi);
        if (lm < 7) {
#pragma unroll
            for (int r = 0; r < 8; ++r) gqM[r + hi * 8][lm] = acc[r];
        }
    }
    __syncthreads();

    // ================= potential backward (VJP) =================
    layer_mulsig<1>(BUF2, BUF0, WRp(ws, 1, 3), 32,  sgp3, wave, lane);
    __syncthreads();
    layer_mulsig<4>(BUF0, BUF1, WRp(ws, 1, 2), 128, sgp2, wave, lane);
    __syncthreads();
    layer_mulsig<4>(BUF1, BUF0, WRp(ws, 1, 1), 128, sgp1, wave, lane);
    __syncthreads();
    if (wave == 0) {
        v8f acc = gemm_tile<4>(BUF0, WRp(ws, 1, 0), 128, lm, lm, hi);
        if (lm < 7) {
#pragma unroll
            for (int r = 0; r < 8; ++r) gqP[r + hi * 8][lm] = acc[r];
        }
    }
    __syncthreads();

    // ========= scalar stage B: rhs assembly + 7x7 SPD Cholesky solve =========
    if (tid < 16) {
        const int m = tid;
        float L[28];
        { int k = 0;
          for (int i = 0; i < 7; ++i)
              for (int j = 0; j <= i; ++j, ++k)
                  L[k] = (i == j) ? fast_exp(cM[m][k]) : cM[m][k]; }
        float A[28];  // lower tri of M + (eps_M + eps_solve) I
        { int k = 0;
          for (int i = 0; i < 7; ++i)
              for (int j = 0; j <= i; ++j, ++k) {
                  float s = (i == j) ? (1e-3f + 1e-4f) : 0.f;
                  for (int t = 0; t <= j; ++t) s += L[TRI(i, t)] * L[TRI(j, t)];
                  A[k] = s;
              } }
        float rhs[7];
        for (int i = 0; i < 7; ++i) rhs[i] = gqM[m][i] - gqP[m][i] - rhsP[m][i];
        // Cholesky with reciprocal diagonal (native v_rcp/v_sqrt, no div-fixup chains)
        float C[28], rinv[7];
        for (int i = 0; i < 7; ++i) {
            for (int j = 0; j < i; ++j) {
                float s = A[TRI(i, j)];
                for (int t = 0; t < j; ++t) s -= C[TRI(i, t)] * C[TRI(j, t)];
                C[TRI(i, j)] = s * rinv[j];
            }
            float s = A[TRI(i, i)];
            for (int t = 0; t < i; ++t) s -= C[TRI(i, t)] * C[TRI(i, t)];
            const float d = __builtin_amdgcn_sqrtf(s);
            C[TRI(i, i)] = d;
            rinv[i] = __builtin_amdgcn_rcpf(d);
        }
        float y[7];
        for (int i = 0; i < 7; ++i) {
            float s = rhs[i];
            for (int t = 0; t < i; ++t) s -= C[TRI(i, t)] * y[t];
            y[i] = s * rinv[i];
        }
        float x[7];
        for (int i = 6; i >= 0; --i) {
            float s = y[i];
            for (int t = i + 1; t < 7; ++t) s -= C[TRI(t, i)] * x[t];
            x[i] = s * rinv[i];
        }
        for (int i = 0; i < 7; ++i) P.out[(size_t)(s0 + m) * 7 + i] = x[i];
    }
}

// ---------------- host launch ----------------
extern "C" void kernel_launch(void* const* d_in, const int* in_sizes, int n_in,
                              void* d_out, int out_size, void* d_ws, size_t ws_size,
                              hipStream_t stream) {
    (void)n_in; (void)out_size; (void)ws_size;
    const float* q  = (const float*)d_in[0];
    const float* qd = (const float*)d_in[1];
    half_t* ws = (half_t*)d_ws;

    static const int od[3] = {28, 1, 28};
    for (int mlp = 0; mlp < 3; ++mlp) {
        for (int l = 0; l < 4; ++l) {
            const float* W = (const float*)d_in[2 + mlp * 8 + 2 * l];
            const int in       = (l == 0) ? 7 : 128;
            const int out      = (l == 3) ? od[mlp] : 128;
            const int in_pad   = (l == 0) ? 32 : 128;
            const int out_pad  = (l == 3) ? 32 : 128;
            const int in_pad16 = (l == 0) ? 16 : 128;
            half_t* wt = ws + mlp * MLP_STRIDE + WT_OFF[l];
            half_t* wr = ws + mlp * MLP_STRIDE + WR_OFF[l];
            const int tot = out_pad * in_pad + in_pad16 * out_pad;
            prep_weights<<<(tot + 255) / 256, 256, 0, stream>>>(W, in, out, in_pad, out_pad,
                                                                in_pad16, wt, wr);
        }
    }

    Params P;
    P.q = q; P.qd = qd; P.ws = ws; P.out = (float*)d_out;
    for (int mlp = 0; mlp < 3; ++mlp)
        for (int l = 0; l < 4; ++l)
            P.b[mlp * 4 + l] = (const float*)d_in[2 + mlp * 8 + 2 * l + 1];

    const int nsamples = in_sizes[0] / 7;       // 65536
    const int nblocks  = nsamples / 16;         // 4096 workgroups of 16 samples
    delan_kernel<<<nblocks, 256, 0, stream>>>(P);
}
